// BPTMemBlock_66279935312383
// MI455X (gfx1250) — matrix operand order repaired and checked by reading, not compiled, not run on hardware
//
#include <hip/hip_runtime.h>
#include <hip/hip_bf16.h>
#include <cstdint>

// ---------------- problem constants (match reference) ----------------
#define NN   65536
#define DD   256
#define HH   8
#define DKK  32
#define DFF_ 1024
#define E1_  1048576
#define E2_  524288
#define SCALE_ 0.17677669529663687f   // 1/sqrt(32)

typedef __attribute__((ext_vector_type(16))) __bf16 v16bf;
typedef __attribute__((ext_vector_type(2)))  __bf16 v2bf;
typedef __attribute__((ext_vector_type(8)))  float  v8f;

// ---------------- helpers ----------------
__device__ __forceinline__ __bf16 f2bf(float f) {
  unsigned u = __float_as_uint(f);
  unsigned r = u + 0x7FFFu + ((u >> 16) & 1u);          // round-to-nearest-even
  unsigned short hs = (unsigned short)(r >> 16);
  return __builtin_bit_cast(__bf16, hs);
}

__device__ __forceinline__ float dot4(float4 a, float4 b) {
  return a.x*b.x + a.y*b.y + a.z*b.z + a.w*b.w;
}

__device__ __forceinline__ void atomicMaxF(float* addr, float v) {
  if (v >= 0.0f) atomicMax((int*)addr, __float_as_int(v));
  else           atomicMin((unsigned int*)addr, __float_as_uint(v));
}

// ---------------- elementwise kernels ----------------
__global__ void k_cvt_bf16(const float* __restrict__ x, __bf16* __restrict__ y, long n) {
  long i = (long)blockIdx.x * blockDim.x + threadIdx.x;
  if (i < n) y[i] = f2bf(x[i]);
}

// W [K,N] row-major f32 -> WT [N,K] row-major bf16 (transposed for B-fragment loads)
__global__ void k_cvt_transpose_bf16(const float* __restrict__ W, __bf16* __restrict__ WT,
                                     int K, int Nc) {
  int idx = blockIdx.x * blockDim.x + threadIdx.x;
  if (idx >= K * Nc) return;
  int kk = idx / Nc, nn = idx % Nc;
  WT[(size_t)nn * K + kk] = f2bf(W[idx]);
}

__global__ void k_fill(float* __restrict__ p, float val, long n) {
  long i = (long)blockIdx.x * blockDim.x + threadIdx.x;
  if (i < n) p[i] = val;
}

// ---------------- WMMA bf16 GEMM: C[M,N] = A[M,K] * B[K,N] (+bias)(+relu) ----------------
// A: bf16 row-major [M,K]; BT: bf16 row-major [N,K] (i.e. B transposed)
// One wave computes a 16(M) x 64(N) strip: one shared A fragment feeds 4 independent
// v_wmma_f32_16x16x32_bf16 accumulators per K-step (hides XDL latency, amortizes A loads).
template <int RELU, int BF16OUT>
__global__ __launch_bounds__(128) void k_gemm_wmma(
    const __bf16* __restrict__ A, const __bf16* __restrict__ BT,
    void* __restrict__ Cout, const float* __restrict__ bias,
    int Mr, int K, int Nc) {
  const int lane = threadIdx.x & 31;
  const int wave = threadIdx.x >> 5;
  const int stripsN = Nc >> 6;                       // N-strips of 64
  const int stripsTotal = (Mr >> 4) * stripsN;
  int strip = blockIdx.x * (blockDim.x >> 5) + wave;
  if (strip >= stripsTotal) return;
  const int tm  = strip / stripsN;
  const int tn4 = strip - tm * stripsN;

  const int hx  = lane >> 4;     // half: 0 = lanes 0-15, 1 = lanes 16-31
  const int l16 = lane & 15;

  const __bf16* Arow = A + (size_t)(tm * 16 + l16) * (size_t)K;
  const __bf16* Brow[4];
  #pragma unroll
  for (int j = 0; j < 4; ++j)
    Brow[j] = BT + (size_t)(tn4 * 64 + j * 16 + l16) * (size_t)K;

  v8f acc0 = {}, acc1 = {}, acc2 = {}, acc3 = {};

  for (int k0 = 0; k0 < K; k0 += 32) {
    if (k0 + 32 < K) {                       // stream prefetch of next K block
      __builtin_prefetch(Arow + k0 + 32, 0, 1);
      __builtin_prefetch(Brow[0] + k0 + 32, 0, 1);
      __builtin_prefetch(Brow[2] + k0 + 32, 0, 1);
    }
    v16bf a;
    // A fragment: VGPR v -> K = k0 + 16*hx + 2v
    #pragma unroll
    for (int v = 0; v < 4; ++v) {
      v2bf p0 = *(const v2bf*)(Arow + k0 + 16*hx + 2*v);
      v2bf p1 = *(const v2bf*)(Arow + k0 + 16*hx + 8 + 2*v);
      a[2*v]   = p0[0]; a[2*v+1]   = p0[1];
      a[8+2*v] = p1[0]; a[8+2*v+1] = p1[1];
    }
    // B fragments: VGPR v -> K = k0 + 16*hx + 2v  (lane = column N%16)
    v16bf b0, b1, b2, b3;
    #pragma unroll
    for (int v = 0; v < 8; ++v) {
      v2bf p0 = *(const v2bf*)(Brow[0] + k0 + 16*hx + 2*v);
      v2bf p1 = *(const v2bf*)(Brow[1] + k0 + 16*hx + 2*v);
      v2bf p2 = *(const v2bf*)(Brow[2] + k0 + 16*hx + 2*v);
      v2bf p3 = *(const v2bf*)(Brow[3] + k0 + 16*hx + 2*v);
      b0[2*v] = p0[0]; b0[2*v+1] = p0[1];
      b1[2*v] = p1[0]; b1[2*v+1] = p1[1];
      b2[2*v] = p2[0]; b2[2*v+1] = p2[1];
      b3[2*v] = p3[0]; b3[2*v+1] = p3[1];
    }
    acc0 = __builtin_amdgcn_wmma_f32_16x16x32_bf16(false, a, false, b0, (short)0, acc0, false, false);
    acc1 = __builtin_amdgcn_wmma_f32_16x16x32_bf16(false, a, false, b1, (short)0, acc1, false, false);
    acc2 = __builtin_amdgcn_wmma_f32_16x16x32_bf16(false, a, false, b2, (short)0, acc2, false, false);
    acc3 = __builtin_amdgcn_wmma_f32_16x16x32_bf16(false, a, false, b3, (short)0, acc3, false, false);
  }

  // epilogue: C layout per tile: lane -> N = lane&15, VGPR v -> M = v + 8*half
  v8f accs[4] = {acc0, acc1, acc2, acc3};
  #pragma unroll
  for (int j = 0; j < 4; ++j) {
    const int cn = tn4 * 64 + j * 16 + l16;
    const float bs = bias ? bias[cn] : 0.0f;
    #pragma unroll
    for (int v = 0; v < 8; ++v) {
      int cm = tm * 16 + 8*hx + v;
      float x = accs[j][v] + bs;
      if (RELU) x = x > 0.0f ? x : 0.0f;
      if (BF16OUT) ((__bf16*)Cout)[(size_t)cm * Nc + cn] = f2bf(x);
      else         ((float* )Cout)[(size_t)cm * Nc + cn] = x;
    }
  }
}

// ---------------- edge kernels (one thread per edge*head) ----------------
// pass 1: e = (k[src].q[dst] + ak[etype].q[dst]) * SCALE ; running segment max
__global__ void k_edge_scores(const float* __restrict__ q, const float* __restrict__ kk,
                              const int* __restrict__ src, const int* __restrict__ dst,
                              const int* __restrict__ etype, const float* __restrict__ ak,
                              float* __restrict__ e, float* __restrict__ mrow, int E) {
  long t = (long)blockIdx.x * blockDim.x + threadIdx.x;
  if (t >= (long)E * HH) return;
  int h  = (int)(t & (HH - 1));
  int ed = (int)(t >> 3);
  int s = src[ed], d = dst[ed];
  const float4* kp = (const float4*)(kk + (size_t)s * DD + h * DKK);
  const float4* qp = (const float4*)(q  + (size_t)d * DD + h * DKK);
  float4 qv[8];
  float acc = 0.0f;
  #pragma unroll
  for (int i = 0; i < 8; ++i) { qv[i] = qp[i]; acc += dot4(kp[i], qv[i]); }
  if (ak) {
    const float4* ap = (const float4*)(ak + (size_t)etype[ed] * DKK);
    float r = 0.0f;
    #pragma unroll
    for (int i = 0; i < 8; ++i) r += dot4(ap[i], qv[i]);
    acc += r;
  }
  acc *= SCALE_;
  e[t] = acc;
  atomicMaxF(&mrow[(size_t)d * HH + h], acc);
}

// pass 2: ex = exp(e - m[dst]); e <- ex; denom[dst] += ex
__global__ void k_edge_expsum(const int* __restrict__ dst, float* __restrict__ e,
                              const float* __restrict__ mrow, float* __restrict__ drow, int E) {
  long t = (long)blockIdx.x * blockDim.x + threadIdx.x;
  if (t >= (long)E * HH) return;
  int h  = (int)(t & (HH - 1));
  int ed = (int)(t >> 3);
  int d  = dst[ed];
  float ex = __expf(e[t] - mrow[(size_t)d * HH + h]);
  e[t] = ex;
  unsafeAtomicAdd(&drow[(size_t)d * HH + h], ex);
}

// pass 3: agg[dst] += (e/denom[dst]) * v[src]
__global__ void k_edge_agg(const int* __restrict__ src, const int* __restrict__ dst,
                           const float* __restrict__ e, const float* __restrict__ drow,
                           const float* __restrict__ v, float* __restrict__ agg, int E) {
  long t = (long)blockIdx.x * blockDim.x + threadIdx.x;
  if (t >= (long)E * HH) return;
  int h  = (int)(t & (HH - 1));
  int ed = (int)(t >> 3);
  int s = src[ed], d = dst[ed];
  float a = e[t] / drow[(size_t)d * HH + h];
  const float* vp = v   + (size_t)s * DD + h * DKK;
  float*       op = agg + (size_t)d * DD + h * DKK;
  #pragma unroll
  for (int i = 0; i < DKK; ++i) unsafeAtomicAdd(op + i, a * vp[i]);
}

// ---------------- residual + layernorm (one wave per 256-elem row) ----------------
__global__ void k_add_ln(const float* __restrict__ x, const float* __restrict__ r,
                         const float* __restrict__ g, const float* __restrict__ b,
                         float* __restrict__ outf, __bf16* __restrict__ outb, int rows) {
  int wave = threadIdx.x >> 5, lane = threadIdx.x & 31;
  int row = blockIdx.x * (blockDim.x >> 5) + wave;
  if (row >= rows) return;
  const float* xp = x + (size_t)row * DD;
  const float* rp = r + (size_t)row * DD;
  float vals[8];
  float s = 0.0f;
  #pragma unroll
  for (int i = 0; i < 8; ++i) { vals[i] = xp[lane + 32*i] + rp[lane + 32*i]; s += vals[i]; }
  #pragma unroll
  for (int off = 16; off >= 1; off >>= 1) s += __shfl_xor(s, off, 32);
  float mu = s * (1.0f / DD);
  float vs = 0.0f;
  #pragma unroll
  for (int i = 0; i < 8; ++i) { float dl = vals[i] - mu; vs += dl * dl; }
  #pragma unroll
  for (int off = 16; off >= 1; off >>= 1) vs += __shfl_xor(vs, off, 32);
  float inv = rsqrtf(vs * (1.0f / DD) + 1e-5f);
  #pragma unroll
  for (int i = 0; i < 8; ++i) {
    int c = lane + 32*i;
    float y = (vals[i] - mu) * inv * g[c] + b[c];
    outf[(size_t)row * DD + c] = y;
    if (outb) outb[(size_t)row * DD + c] = f2bf(y);
  }
}

// ---------------- orchestration ----------------
extern "C" void kernel_launch(void* const* d_in, const int* in_sizes, int n_in,
                              void* d_out, int out_size, void* d_ws, size_t ws_size,
                              hipStream_t stream) {
  const float* h        = (const float*)d_in[0];
  const float* mem      = (const float*)d_in[1];
  const int*   src_intra= (const int*)d_in[2];
  const int*   dst_intra= (const int*)d_in[3];
  const int*   etype    = (const int*)d_in[4];
  const int*   src_inter= (const int*)d_in[5];
  const int*   dst_inter= (const int*)d_in[6];
  const float* Wq0 = (const float*)d_in[7],  *Wk0 = (const float*)d_in[8];
  const float* Wv0 = (const float*)d_in[9],  *Wo0 = (const float*)d_in[10];
  const float* Wq1 = (const float*)d_in[11], *Wk1 = (const float*)d_in[12];
  const float* Wv1 = (const float*)d_in[13], *Wo1 = (const float*)d_in[14];
  const float* ak  = (const float*)d_in[15];
  const float* ln0g = (const float*)d_in[16], *ln0b = (const float*)d_in[17];
  const float* ln1g = (const float*)d_in[18], *ln1b = (const float*)d_in[19];
  const float* ln2g = (const float*)d_in[20], *ln2b = (const float*)d_in[21];
  const float* Wf1 = (const float*)d_in[22], *bf1 = (const float*)d_in[23];
  const float* Wf2 = (const float*)d_in[24], *bf2 = (const float*)d_in[25];

  // ---- workspace carve ----
  char* wp = (char*)d_ws;
  auto carve = [&](size_t bytes) -> void* {
    void* p = wp; wp += (bytes + 255) & ~(size_t)255; return p;
  };
  const size_t nd = (size_t)NN * DD;
  __bf16* hb   = (__bf16*)carve(nd * 2);          // bf16 activations (A operand)
  __bf16* memb = (__bf16*)carve(nd * 2);
  __bf16* ab   = (__bf16*)carve(nd * 2);          // bf16 of attention aggregate
  __bf16* WT[8];
  for (int i = 0; i < 8; ++i) WT[i] = (__bf16*)carve((size_t)DD * DD * 2);
  __bf16* Wf1T = (__bf16*)carve((size_t)DD * DFF_ * 2);
  __bf16* Wf2T = (__bf16*)carve((size_t)DFF_ * DD * 2);
  float* qb   = (float*)carve(nd * 4);
  float* kb   = (float*)carve(nd * 4);
  float* vb   = (float*)carve(nd * 4);
  float* agg  = (float*)carve(nd * 4);
  float* proj = (float*)carve(nd * 4);
  float* h1   = (float*)carve(nd * 4);
  float* h2   = (float*)carve(nd * 4);
  float* ebuf = (float*)carve((size_t)E1_ * HH * 4);
  float* mrow = (float*)carve((size_t)NN * HH * 4);
  float* drow = (float*)carve((size_t)NN * HH * 4);
  __bf16* ffi = (__bf16*)carve((size_t)NN * DFF_ * 2);

  auto blks = [](long n, int t) { return (unsigned)((n + t - 1) / t); };

  auto gemmF = [&](const __bf16* A, const __bf16* BT, float* C, const float* bias,
                   int Mr, int K, int Nc) {
    int strips = (Mr / 16) * (Nc / 64);
    k_gemm_wmma<0,0><<<blks(strips, 4), 128, 0, stream>>>(A, BT, (void*)C, bias, Mr, K, Nc);
  };

  // ---- weight / input conversion ----
  k_cvt_bf16<<<blks(nd,256),256,0,stream>>>(h,   hb,   nd);
  k_cvt_bf16<<<blks(nd,256),256,0,stream>>>(mem, memb, nd);
  const float* Ws[8] = {Wq0, Wk0, Wv0, Wo0, Wq1, Wk1, Wv1, Wo1};
  for (int i = 0; i < 8; ++i)
    k_cvt_transpose_bf16<<<blks(DD*DD,256),256,0,stream>>>(Ws[i], WT[i], DD, DD);
  k_cvt_transpose_bf16<<<blks(DD*DFF_,256),256,0,stream>>>(Wf1, Wf1T, DD, DFF_);
  k_cvt_transpose_bf16<<<blks(DFF_*DD,256),256,0,stream>>>(Wf2, Wf2T, DFF_, DD);

  // ================= intra attention =================
  gemmF(hb, WT[0], qb, nullptr, NN, DD, DD);
  gemmF(hb, WT[1], kb, nullptr, NN, DD, DD);
  gemmF(hb, WT[2], vb, nullptr, NN, DD, DD);
  k_fill<<<blks((long)NN*HH,256),256,0,stream>>>(mrow, -3.402823466e38f, (long)NN*HH);
  k_fill<<<blks((long)NN*HH,256),256,0,stream>>>(drow, 0.0f, (long)NN*HH);
  k_fill<<<blks((long)nd,256),256,0,stream>>>(agg, 0.0f, (long)nd);
  k_edge_scores<<<blks((long)E1_*HH,256),256,0,stream>>>(qb, kb, src_intra, dst_intra,
                                                         etype, ak, ebuf, mrow, E1_);
  k_edge_expsum<<<blks((long)E1_*HH,256),256,0,stream>>>(dst_intra, ebuf, mrow, drow, E1_);
  k_edge_agg<<<blks((long)E1_*HH,256),256,0,stream>>>(src_intra, dst_intra, ebuf, drow,
                                                      vb, agg, E1_);
  k_cvt_bf16<<<blks(nd,256),256,0,stream>>>(agg, ab, nd);
  gemmF(ab, WT[3], proj, nullptr, NN, DD, DD);
  k_add_ln<<<blks((long)NN,8), 256, 0, stream>>>(h, proj, ln0g, ln0b, h1, hb, NN);

  // ================= inter (memory) attention =================
  gemmF(hb,   WT[4], qb, nullptr, NN, DD, DD);
  gemmF(memb, WT[5], kb, nullptr, NN, DD, DD);
  gemmF(memb, WT[6], vb, nullptr, NN, DD, DD);
  k_fill<<<blks((long)NN*HH,256),256,0,stream>>>(mrow, -3.402823466e38f, (long)NN*HH);
  k_fill<<<blks((long)NN*HH,256),256,0,stream>>>(drow, 0.0f, (long)NN*HH);
  k_fill<<<blks((long)nd,256),256,0,stream>>>(agg, 0.0f, (long)nd);
  k_edge_scores<<<blks((long)E2_*HH,256),256,0,stream>>>(qb, kb, src_inter, dst_inter,
                                                         nullptr, nullptr, ebuf, mrow, E2_);
  k_edge_expsum<<<blks((long)E2_*HH,256),256,0,stream>>>(dst_inter, ebuf, mrow, drow, E2_);
  k_edge_agg<<<blks((long)E2_*HH,256),256,0,stream>>>(src_inter, dst_inter, ebuf, drow,
                                                      vb, agg, E2_);
  k_cvt_bf16<<<blks(nd,256),256,0,stream>>>(agg, ab, nd);
  gemmF(ab, WT[7], proj, nullptr, NN, DD, DD);
  k_add_ln<<<blks((long)NN,8), 256, 0, stream>>>(h1, proj, ln1g, ln1b, h2, hb, NN);

  // ================= FFN =================
  {
    int strips = (NN / 16) * (DFF_ / 64);
    k_gemm_wmma<1,1><<<blks(strips,4),128,0,stream>>>(hb, Wf1T, (void*)ffi, bf1,
                                                      NN, DD, DFF_);
  }
  gemmF(ffi, Wf2T, proj, bf2, NN, DFF_, DD);
  k_add_ln<<<blks((long)NN,8), 256, 0, stream>>>(h2, proj, ln2g, ln2b,
                                                 (float*)d_out, nullptr, NN);
}
